// CrossAttention_84765474554552
// MI455X (gfx1250) — matrix operand order repaired and checked by reading, not compiled, not run on hardware
//
#include <hip/hip_runtime.h>

typedef __attribute__((ext_vector_type(16))) _Float16 v16h;
typedef __attribute__((ext_vector_type(8)))  _Float16 v8h;
typedef __attribute__((ext_vector_type(8)))  float    v8f;
typedef __attribute__((ext_vector_type(4)))  int      v4i;

typedef __attribute__((address_space(1))) v4i gv4i;   // global v4i
typedef __attribute__((address_space(3))) v4i lv4i;   // LDS v4i

#if defined(__gfx1250__) && __has_builtin(__builtin_amdgcn_global_load_async_to_lds_b128) && \
    __has_builtin(__builtin_amdgcn_s_wait_asynccnt)
#define HAVE_ASYNC_LDS 1
#else
#define HAVE_ASYNC_LDS 0
#endif

static __device__ __forceinline__ v16h combine16(v8h lo, v8h hi) {
    v16h r;
#pragma unroll
    for (int i = 0; i < 8; ++i) { r[i] = lo[i]; r[i + 8] = hi[i]; }
    return r;
}

// Copy 16 bytes global -> LDS: async DMA path on CDNA5, VGPR-staged fallback.
static __device__ __forceinline__ void copy16_g2l(const _Float16* g, _Float16* l) {
#if HAVE_ASYNC_LDS
    __builtin_amdgcn_global_load_async_to_lds_b128((gv4i*)g, (lv4i*)l, 0, 0);
#else
    *(v8h*)l = *(const v8h*)g;
#endif
}
static __device__ __forceinline__ void async_join() {
#if HAVE_ASYNC_LDS
    __builtin_amdgcn_s_wait_asynccnt(0);
#endif
}

// ---------------------------------------------------------------------------
// f32 -> f16 conversion
// ---------------------------------------------------------------------------
__global__ void cvt_f32_f16(const float* __restrict__ s, _Float16* __restrict__ d, long n) {
    long i = (long)blockIdx.x * blockDim.x + threadIdx.x;
    if (i < n) d[i] = (_Float16)s[i];
}

// ---------------------------------------------------------------------------
// RMS-norm one row of D floats, emit f16
// ---------------------------------------------------------------------------
__global__ __launch_bounds__(256) void rmsnorm_f16(const float* __restrict__ src,
                                                   _Float16* __restrict__ dst, int D) {
    __shared__ float red[8];
    int row = blockIdx.x;
    const float* p = src + (size_t)row * D;
    float ss = 0.f;
    for (int i = threadIdx.x; i < D; i += 256) { float v = p[i]; ss += v * v; }
#pragma unroll
    for (int m = 1; m < 32; m <<= 1) ss += __shfl_xor(ss, m, 32);
    int w = threadIdx.x >> 5, lane = threadIdx.x & 31;
    if (lane == 0) red[w] = ss;
    __syncthreads();
    if (threadIdx.x < 8) {
        float v = red[threadIdx.x];
#pragma unroll
        for (int m = 1; m < 8; m <<= 1) v += __shfl_xor(v, m, 32);
        if (threadIdx.x == 0) red[0] = v;
    }
    __syncthreads();
    float inv = rsqrtf(red[0] / (float)D + 1e-6f);
    _Float16* q = dst + (size_t)row * D;
    for (int i = threadIdx.x; i < D; i += 256) q[i] = (_Float16)(p[i] * inv);
}

// ---------------------------------------------------------------------------
// Generic f16 GEMM with bias: C(MxN) = A(MxK) * B(KxN) + bias
// 128x128 tile per 256-thread block (8 waves), WMMA f32_16x16x32_f16.
// M,N multiples of 128; K multiple of 32. OUT_F16 chosen at compile time.
// ---------------------------------------------------------------------------
#define GT_M 128
#define GT_N 128
#define GT_K 32

template <int OUT_F16>
__global__ __launch_bounds__(256) void gemm_f16_wmma(
    const _Float16* __restrict__ A, const _Float16* __restrict__ Bw,
    const float* __restrict__ bias, void* __restrict__ C,
    int M, int N, int K) {
    __shared__ _Float16 lA[GT_M * GT_K];   // [m][k]
    __shared__ _Float16 lBt[GT_N * GT_K];  // [n][k] (transposed)

    int tid = threadIdx.x;
    int m0 = blockIdx.y * GT_M;
    int n0 = blockIdx.x * GT_N;
    int w = tid >> 5, lane = tid & 31;
    int wm = w >> 2, wn = w & 3;            // wave tile: rows wm*64..+63, cols wn*32..+31
    int lm = lane & 15;
    int half = lane >> 4;
    int kbA = half * 8;                     // A-operand K base (split 8-chunks)
    int kbB = half * 8;                     // B-operand K base (split 8-chunks)

    v8f zf = {};
    v8f acc[4][2];
#pragma unroll
    for (int i = 0; i < 4; ++i)
#pragma unroll
        for (int j = 0; j < 2; ++j) acc[i][j] = zf;

    int arow = tid >> 1, aseg = (tid & 1) * 16;   // A coop load: 128 rows x 2 segs
    int brow = tid >> 3, bseg = (tid & 7) * 16;   // B coop load: 32 rows x 8 segs

    for (int kk = 0; kk < K; kk += GT_K) {
        // A tile: async DMA straight into LDS (two b128 per thread)
        const _Float16* ag = A + (size_t)(m0 + arow) * K + kk + aseg;
        copy16_g2l(ag,     &lA[arow * GT_K + aseg]);
        copy16_g2l(ag + 8, &lA[arow * GT_K + aseg + 8]);

        // B tile: transpose during store (scalar b16 stores)
        const _Float16* bp = Bw + (size_t)(kk + brow) * N + n0 + bseg;
#pragma unroll
        for (int e = 0; e < 16; ++e) lBt[(bseg + e) * GT_K + brow] = bp[e];
        async_join();
        __syncthreads();

        v16h afr[4], bfr[2];
#pragma unroll
        for (int i = 0; i < 4; ++i) {
            int r = wm * 64 + i * 16 + lm;
            afr[i] = combine16(*(const v8h*)(&lA[r * GT_K + kbA]),
                               *(const v8h*)(&lA[r * GT_K + kbA + 16]));
        }
#pragma unroll
        for (int j = 0; j < 2; ++j) {
            int c = wn * 32 + j * 16 + lm;
            bfr[j] = combine16(*(const v8h*)(&lBt[c * GT_K + kbB]),
                               *(const v8h*)(&lBt[c * GT_K + kbB + 16]));
        }
#pragma unroll
        for (int i = 0; i < 4; ++i)
#pragma unroll
            for (int j = 0; j < 2; ++j)
                acc[i][j] = __builtin_amdgcn_wmma_f32_16x16x32_f16(
                    false, afr[i], false, bfr[j], (short)0, acc[i][j], false, false);
        __syncthreads();
    }

    // Branch-free epilogue: C layout -> element (m = r + half*8, n = lane&15)
#pragma unroll
    for (int i = 0; i < 4; ++i) {
#pragma unroll
        for (int j = 0; j < 2; ++j) {
            int col = n0 + wn * 32 + j * 16 + lm;
            float bv = bias[col];
            int row0 = m0 + wm * 64 + i * 16 + half * 8;
#pragma unroll
            for (int r = 0; r < 8; ++r) {
                float v = acc[i][j][r] + bv;
                if constexpr (OUT_F16)
                    ((_Float16*)C)[(size_t)(row0 + r) * N + col] = (_Float16)v;
                else
                    ((float*)C)[(size_t)(row0 + r) * N + col] = v;
            }
        }
    }
}

// ---------------------------------------------------------------------------
// Cross-attention core. Grid: (N/64, H, B); 128 threads (4 waves x 16 q-rows).
// Q:(B*N,1024) f16, col = h*64+hd.  KV:(B*Kc,2048): K at col h*64+hd, V at +1024.
// O:(B*N,1024) f16. logits clipped to [-10,10] => exp() bounded, no running max
// needed (softmax is shift-invariant; exp(10)*2048 fits f32 comfortably).
// ---------------------------------------------------------------------------
__global__ __launch_bounds__(128) void attn_wmma(
    const _Float16* __restrict__ Q, const _Float16* __restrict__ KV,
    _Float16* __restrict__ O, int Bq, int Nq, int Kc) {
    const int Hd = 64, Dq = 1024, Dkv = 2048;
    __shared__ _Float16 lK[32 * 64];       // [key][dim]
    __shared__ _Float16 lV[64 * 32];       // [dim][key] (transposed)
    __shared__ _Float16 lS[4][16 * 32];    // per-wave P tile [qrow][key]

    int tid = threadIdx.x;
    int w = tid >> 5, lane = tid & 31;
    int lm = lane & 15;
    int half = lane >> 4;
    int kbA = half * 8;
    int kbB = half * 8;
    int h = blockIdx.y;
    int b = blockIdx.z;
    int q0 = blockIdx.x * 64 + w * 16;     // this wave's 16 q-rows

    // Q fragments (A operand), pre-scaled by Hd^-0.5 (exact in f16)
    v16h qfr[2];
#pragma unroll
    for (int kh = 0; kh < 2; ++kh) {
        const _Float16* qp = Q + (size_t)(b * Nq + q0 + lm) * Dq + h * Hd + kh * 32 + kbA;
        v16h t = combine16(*(const v8h*)qp, *(const v8h*)(qp + 16));
#pragma unroll
        for (int e = 0; e < 16; ++e) t[e] = t[e] * (_Float16)0.125f;
        qfr[kh] = t;
    }

    v8f zf = {};
    v8f oacc[4];
#pragma unroll
    for (int d = 0; d < 4; ++d) oacc[d] = zf;
    float lsum[8];
#pragma unroll
    for (int r = 0; r < 8; ++r) lsum[r] = 0.f;

    int key = tid >> 2;                    // 0..31 (coop loads)
    int seg = (tid & 3) * 16;              // dim segment

    for (int kt = 0; kt < Kc; kt += 32) {
        const _Float16* kp = KV + (size_t)(b * Kc + kt + key) * Dkv + h * Hd + seg;
        // K tile: async DMA to LDS
        copy16_g2l(kp,     &lK[key * 64 + seg]);
        copy16_g2l(kp + 8, &lK[key * 64 + seg + 8]);
        // V tile: transpose during store
        const _Float16* vp = kp + 1024;
#pragma unroll
        for (int e = 0; e < 16; ++e) lV[(seg + e) * 32 + key] = vp[e];
        async_join();
        __syncthreads();

        // S(16q x 32keys) = Qs * K^T
        v8f s[2];
        s[0] = zf; s[1] = zf;
#pragma unroll
        for (int kblk = 0; kblk < 2; ++kblk) {
            int krow = kblk * 16 + lm;     // B-operand lane = key column
#pragma unroll
            for (int kh = 0; kh < 2; ++kh) {
                v16h kf = combine16(*(const v8h*)(&lK[krow * 64 + kh * 32 + kbB]),
                                    *(const v8h*)(&lK[krow * 64 + kh * 32 + kbB + 16]));
                s[kblk] = __builtin_amdgcn_wmma_f32_16x16x32_f16(
                    false, qfr[kh], false, kf, (short)0, s[kblk], false, false);
            }
        }
        // clip, exp; accumulate partial row sums; stage P into LDS
#pragma unroll
        for (int kblk = 0; kblk < 2; ++kblk) {
#pragma unroll
            for (int r = 0; r < 8; ++r) {
                float sv = fminf(fmaxf(s[kblk][r], -10.f), 10.f);
                float p = __expf(sv);
                lsum[r] += p;
                int mm = r + half * 8;
                lS[w][mm * 32 + kblk * 16 + lm] = (_Float16)p;
            }
        }
        // O(16q x 64d) += P(16x32) * V(32x64)
        v16h pfr = combine16(*(const v8h*)(&lS[w][lm * 32 + kbA]),
                             *(const v8h*)(&lS[w][lm * 32 + kbA + 16]));
#pragma unroll
        for (int d = 0; d < 4; ++d) {
            int dim = d * 16 + lm;
            v16h vf = combine16(*(const v8h*)(&lV[dim * 32 + kbB]),
                                *(const v8h*)(&lV[dim * 32 + kbB + 16]));
            oacc[d] = __builtin_amdgcn_wmma_f32_16x16x32_f16(
                false, pfr, false, vf, (short)0, oacc[d], false, false);
        }
        __syncthreads();
    }

    // row-sum reduce over the 16 lanes of each half, normalize, store
#pragma unroll
    for (int r = 0; r < 8; ++r) {
        float sx = lsum[r];
#pragma unroll
        for (int m = 1; m < 16; m <<= 1) sx += __shfl_xor(sx, m, 32);
        lsum[r] = 1.f / sx;
    }
#pragma unroll
    for (int d = 0; d < 4; ++d) {
#pragma unroll
        for (int r = 0; r < 8; ++r) {
            int mm = r + half * 8;
            int row = b * Nq + q0 + mm;
            int col = h * Hd + d * 16 + lm;
            O[(size_t)row * Dq + col] = (_Float16)(oacc[d][r] * lsum[r]);
        }
    }
}

// attn.mean() == 1/K exactly (softmax rows sum to 1)
__global__ void write_mean(float* out, long idx, float v) {
    if (threadIdx.x == 0 && blockIdx.x == 0) out[idx] = v;
}

// ---------------------------------------------------------------------------
extern "C" void kernel_launch(void* const* d_in, const int* in_sizes, int n_in,
                              void* d_out, int out_size, void* d_ws, size_t ws_size,
                              hipStream_t stream) {
    const float* x      = (const float*)d_in[0];
    const float* ctx    = (const float*)d_in[1];
    const float* q_w    = (const float*)d_in[2];
    const float* q_b    = (const float*)d_in[3];
    const float* kv_w   = (const float*)d_in[4];
    const float* kv_b   = (const float*)d_in[5];
    const float* proj_w = (const float*)d_in[6];
    const float* proj_b = (const float*)d_in[7];
    float* out = (float*)d_out;

    const int B = 4, N = 2048, Kc = 2048, D = 1024, H = 16;
    const long MN = (long)B * N;   // 8192 rows (== B*Kc too)

    _Float16* ws16 = (_Float16*)d_ws;
    _Float16* xh   = ws16;                         // 8M halfs
    _Float16* ch   = xh   + (size_t)MN * D;        // 8M
    _Float16* wqh  = ch   + (size_t)MN * D;        // 1M
    _Float16* wkvh = wqh  + (size_t)D * D;         // 2M
    _Float16* wph  = wkvh + (size_t)D * 2 * D;     // 1M
    _Float16* Qh   = wph  + (size_t)D * D;         // 8M
    _Float16* KVh  = Qh   + (size_t)MN * D;        // 16M
    _Float16* Oh   = xh;                           // reuse x-staging after Q GEMM

    long nx = MN * (long)D;
    cvt_f32_f16<<<dim3((nx + 255) / 256), dim3(256), 0, stream>>>(x, xh, nx);
    cvt_f32_f16<<<dim3(((long)D * D + 255) / 256), dim3(256), 0, stream>>>(q_w, wqh, (long)D * D);
    cvt_f32_f16<<<dim3(((long)D * 2 * D + 255) / 256), dim3(256), 0, stream>>>(kv_w, wkvh, (long)D * 2 * D);
    cvt_f32_f16<<<dim3(((long)D * D + 255) / 256), dim3(256), 0, stream>>>(proj_w, wph, (long)D * D);
    rmsnorm_f16<<<dim3(B * Kc), dim3(256), 0, stream>>>(ctx, ch, D);

    // Q = x @ q_w + q_b          (8192 x 1024 x 1024) -> f16
    gemm_f16_wmma<1><<<dim3(D / GT_N, MN / GT_M), dim3(256), 0, stream>>>(
        xh, wqh, q_b, Qh, (int)MN, D, D);
    // KV = ctxn @ kv_w + kv_b    (8192 x 2048 x 1024) -> f16
    gemm_f16_wmma<1><<<dim3(2 * D / GT_N, MN / GT_M), dim3(256), 0, stream>>>(
        ch, wkvh, kv_b, KVh, (int)MN, 2 * D, D);
    // attention
    attn_wmma<<<dim3(N / 64, H, B), dim3(128), 0, stream>>>(Qh, KVh, Oh, B, N, Kc);
    // out = O @ proj_w + proj_b  (8192 x 1024 x 1024) -> f32 into d_out
    gemm_f16_wmma<0><<<dim3(D / GT_N, MN / GT_M), dim3(256), 0, stream>>>(
        Oh, wph, proj_b, out, (int)MN, D, D);
    // trailing scalar: attn.mean() == 1/Kc
    write_mean<<<dim3(1), dim3(1), 0, stream>>>(out, nx, 1.0f / (float)Kc);
}
